// Model_32676111188683
// MI455X (gfx1250) — matrix operand order repaired and checked by reading, not compiled, not run on hardware
//
#include <hip/hip_runtime.h>
#include <hip/hip_bf16.h>

// ---------------------------------------------------------------------------
// 3-layer tanh RNN, persistent single-workgroup kernel for MI455X (gfx1250).
//
//   B=64, T=1024, IN=2, H=512, L=3
//
// The recurrence serializes 3072 layer-steps (each a 64x512 GEMM, K=512/1024),
// so one persistent workgroup (32 wave32s on one WGP) holds h[3][64][512] in
// LDS (bf16, padded rows -> 195 KB of the 320 KB WGP LDS) and streams bf16
// weights (2.5 MB, L2-resident) with v_wmma_f32_16x16x32_bf16.
//
// Tile map: wave w owns N-tile n0=w*16 and ALL four M-tiles, so every weight
// byte crosses L2->WGP exactly once per step (no cross-wave B duplication);
// the A fan-out is served from LDS. h rows are padded by 16 B so the 16 rows
// of an A-tile ds_load_b128 land on disjoint bank groups (min 2-cycle access).
// Per K-step, ALL operand loads (1 B + 4 A tiles) are issued before the four
// WMMAs so the split wait counters can drain progressively instead of gating
// each WMMA on an s_wait_dscnt 0x0.
// ---------------------------------------------------------------------------

#define B_ 64
#define T_ 1024
#define H_ 512
#define L_ 3
#define HP_ 520   // padded LDS row stride in bf16 elements (+16 B): 260 dwords
                  // per row => rows step 4 banks => conflict-minimal A loads

typedef __attribute__((ext_vector_type(16))) __bf16 v16bf;
typedef __attribute__((ext_vector_type(8)))  __bf16 v8bf;
typedef __attribute__((ext_vector_type(8)))  float  v8f;

#define WMMA_BF16(a, b, c) \
    __builtin_amdgcn_wmma_f32_16x16x32_bf16(false, (a), false, (b), (short)0, (c), false, false)

// Build a 16-wide bf16 A operand from two 16-byte LDS chunks.
// Lane<16:  elems 0-7 = K kb+0..7,  elems 8-15 = K kb+16..23
// Lane>=16: elems 0-7 = K kb+8..15, elems 8-15 = K kb+24..31  (p includes hi*8)
__device__ __forceinline__ v16bf load_a_pair(const __bf16* p) {
    v8bf lo = *(const v8bf*)(p);
    v8bf hi = *(const v8bf*)(p + 16);
    return __builtin_shufflevector(lo, hi, 0, 1, 2, 3, 4, 5, 6, 7,
                                           8, 9, 10, 11, 12, 13, 14, 15);
}

// c[mt] += A(mt-th 16-row block of hbase) * w^T for this wave's 16 columns.
//   hbase : LDS base of this layer's 64xH_ state (row stride HP_)
//   w     : bf16 weight matrix, row-major [H_][H_]  (B[k][n] = w[n][k])
//   n0    : this wave's output-column base (one 16-wide N-tile)
__device__ __forceinline__ void gemm_acc(v8f* c, const __bf16* __restrict__ hbase,
                                         const __bf16* __restrict__ w,
                                         int n0, int lr, int hi) {
    // B layout (32x16 bf16): lane 0-15 = col n0+lane, K=kb+0..7,kb+16..23; lane 16-31 =
    // col n0+lane-16, K=kb+8..15,kb+24..31  -> two 16 B loads per lane (same as A).
    const __bf16* wr = w + (size_t)(n0 + lr) * H_ + hi * 8;
    const __bf16* a0 = hbase + (size_t)( 0 + lr) * HP_ + hi * 8;
    const __bf16* a1 = hbase + (size_t)(16 + lr) * HP_ + hi * 8;
    const __bf16* a2 = hbase + (size_t)(32 + lr) * HP_ + hi * 8;
    const __bf16* a3 = hbase + (size_t)(48 + lr) * HP_ + hi * 8;

    for (int kb = 0; kb < H_; kb += 32) {
        // Issue every operand load before the first WMMA: one clause of global
        // B128s for B, eight ds_load_b128 for the A tiles. The WMMAs then
        // retire against progressively draining LOADcnt/DScnt instead of a
        // full-stop wait per matrix op.
        v16bf b  = load_a_pair(wr + kb);                 // single B stream
        __builtin_prefetch(wr + kb + 128, 0, 1);         // 256 B (4 K-steps) ahead
        v16bf x0 = load_a_pair(a0 + kb);
        v16bf x1 = load_a_pair(a1 + kb);
        v16bf x2 = load_a_pair(a2 + kb);
        v16bf x3 = load_a_pair(a3 + kb);
        c[0] = WMMA_BF16(x0, b, c[0]);
        c[1] = WMMA_BF16(x1, b, c[1]);
        c[2] = WMMA_BF16(x2, b, c[2]);
        c[3] = WMMA_BF16(x3, b, c[3]);
    }
}

// One-time fp32 -> bf16 weight conversion into workspace (amortized over T).
__global__ void cvt_weights(const float* __restrict__ wih_rest,   // [2][512][512]
                            const float* __restrict__ whh,        // [3][512][512]
                            __bf16* __restrict__ wih_bf,
                            __bf16* __restrict__ whh_bf) {
    int i = blockIdx.x * blockDim.x + threadIdx.x;
    if (i < 2 * H_ * H_) wih_bf[i] = (__bf16)wih_rest[i];
    if (i < 3 * H_ * H_) whh_bf[i] = (__bf16)whh[i];
}

__global__ __launch_bounds__(1024, 1)
void rnn_persistent(const float* __restrict__ input,      // [B,T,IN]
                    const float* __restrict__ hidden,     // [L,B,H]
                    const float* __restrict__ W_ih0,      // [H,2]
                    const float* __restrict__ b_ih0,      // [H]
                    const float* __restrict__ b_ih_rest,  // [L-1,H]
                    const float* __restrict__ b_hh,       // [L,H]
                    const float* __restrict__ W_dense,    // [2,H]
                    const float* __restrict__ b_dense,    // [2]
                    const __bf16* __restrict__ wih_bf,    // [2][H][H]
                    const __bf16* __restrict__ whh_bf,    // [3][H][H]
                    float* __restrict__ out) {            // [128] ++ [L,B,H]
    extern __shared__ char smem[];
    __bf16* h  = (__bf16*)smem;                        // [L][B][HP_] state
    float*  xs = (float*)(smem + L_ * B_ * HP_ * 2);   // [B][2] current x_t

    const int tid  = threadIdx.x;
    const int lane = tid & 31;
    const int wave = tid >> 5;       // 0..31
    const int lr   = lane & 15;
    const int hi   = lane >> 4;      // K-half select
    const int n0   = wave * 16;      // this wave's single N-tile

    // Init state from `hidden` (strip into padded rows).
    for (int i = tid; i < L_ * B_ * H_; i += 1024) {
        int r = i >> 9, n = i & (H_ - 1);
        h[(size_t)r * HP_ + n] = (__bf16)hidden[i];
    }
    __syncthreads();

    for (int t = 0; t < T_; ++t) {
        // Stage x_t (batch-first input layout).
        if (tid < B_ * 2)
            xs[tid] = input[(size_t)(tid >> 1) * (T_ * 2) + t * 2 + (tid & 1)];
        __syncthreads();

        for (int l = 0; l < L_; ++l) {
            const int n = n0 + lr;   // same output column for all 4 M-tiles
            v8f c[4];
            // Accumulator init: biases (+ K=2 input projection for layer 0 in
            // VALU — a GEMM would waste the K=32 WMMA depth on K=2).
            {
                float bias = b_hh[l * H_ + n] +
                             (l == 0 ? b_ih0[n] : b_ih_rest[(l - 1) * H_ + n]);
                float w0 = 0.f, w1 = 0.f;
                if (l == 0) { w0 = W_ih0[n * 2 + 0]; w1 = W_ih0[n * 2 + 1]; }
                #pragma unroll
                for (int mt = 0; mt < 4; ++mt) {
                    #pragma unroll
                    for (int e = 0; e < 8; ++e) {
                        const int m = mt * 16 + e + hi * 8;  // C/D element row
                        float v = bias;
                        if (l == 0) v += xs[m * 2 + 0] * w0 + xs[m * 2 + 1] * w1;
                        c[mt][e] = v;
                    }
                }
            }

            // h_l(t-1) @ Whh_l^T
            gemm_acc(c, h + (size_t)(l * B_) * HP_,
                     whh_bf + (size_t)l * H_ * H_, n0, lr, hi);

            // h_{l-1}(t) @ Wih_l^T (layer below already updated this t)
            if (l > 0)
                gemm_acc(c, h + (size_t)((l - 1) * B_) * HP_,
                         wih_bf + (size_t)(l - 1) * H_ * H_, n0, lr, hi);

            __syncthreads();  // all waves done reading h[l](t-1) before overwrite

            #pragma unroll
            for (int mt = 0; mt < 4; ++mt) {
                #pragma unroll
                for (int e = 0; e < 8; ++e) {
                    const int m = mt * 16 + e + hi * 8;
                    h[(size_t)(l * B_ + m) * HP_ + n] = (__bf16)tanhf(c[mt][e]);
                }
            }
            __syncthreads();  // h[l](t) visible to layer l+1 / next t
        }
    }

    // h_finals -> out[128 ..)
    for (int i = tid; i < L_ * B_ * H_; i += 1024) {
        int r = i >> 9, n = i & (H_ - 1);
        out[128 + i] = (float)h[(size_t)r * HP_ + n];
    }
    // Dense head on top layer's final state: out[b*2+o], K=512 dot products.
    if (tid < B_ * 2) {
        const int b = tid >> 1, o = tid & 1;
        float s = b_dense[o];
        const __bf16* hrow = h + (size_t)(2 * B_ + b) * HP_;
        for (int k = 0; k < H_; ++k)
            s += (float)hrow[k] * W_dense[o * H_ + k];
        out[b * 2 + o] = s;
    }
}

extern "C" void kernel_launch(void* const* d_in, const int* in_sizes, int n_in,
                              void* d_out, int out_size, void* d_ws, size_t ws_size,
                              hipStream_t stream) {
    (void)in_sizes; (void)n_in; (void)out_size; (void)ws_size;

    const float* input     = (const float*)d_in[0];
    const float* hidden    = (const float*)d_in[1];
    const float* W_ih0     = (const float*)d_in[2];
    const float* b_ih0     = (const float*)d_in[3];
    const float* W_ih_rest = (const float*)d_in[4];
    const float* b_ih_rest = (const float*)d_in[5];
    const float* W_hh      = (const float*)d_in[6];
    const float* b_hh      = (const float*)d_in[7];
    const float* W_dense   = (const float*)d_in[8];
    const float* b_dense   = (const float*)d_in[9];
    float* out = (float*)d_out;

    // Workspace: bf16 weights (2.5 MB total, stays L2-resident).
    __bf16* wih_bf = (__bf16*)d_ws;                       // 2*512*512
    __bf16* whh_bf = wih_bf + 2 * H_ * H_;                // 3*512*512

    const int ncvt = 3 * H_ * H_;
    cvt_weights<<<(ncvt + 255) / 256, 256, 0, stream>>>(W_ih_rest, W_hh,
                                                        wih_bf, whh_bf);

    const size_t smem = (size_t)L_ * B_ * HP_ * sizeof(__bf16)  // ~195 KB state
                      + (size_t)B_ * 2 * sizeof(float);         // x_t stage
    // Opt in to >64KB dynamic LDS (gfx1250 WGP has 320KB).
    (void)hipFuncSetAttribute((const void*)rnn_persistent,
                              hipFuncAttributeMaxDynamicSharedMemorySize,
                              (int)smem);

    rnn_persistent<<<1, 1024, smem, stream>>>(input, hidden, W_ih0, b_ih0,
                                              b_ih_rest, b_hh, W_dense, b_dense,
                                              wih_bf, whh_bf, out);
}
